// SwinTransformer_850403524731
// MI455X (gfx1250) — matrix operand order repaired and checked by reading, not compiled, not run on hardware
//
#include <hip/hip_runtime.h>
#include <hip/hip_fp16.h>

// MI455X / gfx1250, wave32. All GEMMs via v_wmma_f32_16x16x32_f16.

typedef __attribute__((ext_vector_type(16))) _Float16 v16h;
typedef __attribute__((ext_vector_type(8)))  float    v8f;

#define LTOK 262144              // 64^3 tokens per batch
#define NTOK 524288              // B * LTOK

__device__ inline v8f wmma32(v16h a, v16h b, v8f c) {
  return __builtin_amdgcn_wmma_f32_16x16x32_f16(false, a, false, b, (short)0, c, false, false);
}

// Loads one 16-half fragment: halves [0..7] at p, [8..15] at p+16 (A layout),
// for A only; prepacked B fragments are read directly as 16 contiguous halves.
__device__ inline v16h load_frag16(const _Float16* p) {
  union { v16h v; float4 f[2]; } u;
  u.f[0] = *reinterpret_cast<const float4*>(p);
  u.f[1] = *reinterpret_cast<const float4*>(p + 16);
  return u.v;
}

// Bug-faithful relative-position index: idx(h,q) = 3*(c0h-c0q+1)+(c1h-c1q+1)+(c2h-c2q+1),
// where for token u: c0 = bit1(u), c1 = bit2(u), c2 = bit0(u) (numpy meshgrid 'xy').
__device__ inline int rel_idx(int u1, int u2) {
  int c0a = (u1 >> 1) & 1, c1a = (u1 >> 2) & 1, c2a = u1 & 1;
  int c0b = (u2 >> 1) & 1, c1b = (u2 >> 2) & 1, c2b = u2 & 1;
  return 3 * (c0a - c0b + 1) + (c1a - c1b + 1) + (c2a - c2b + 1);
}

// Shift-mask region counter for a token of window (zi,yi,xi): regions exist only
// at the last window along each axis (z==62 -> 1, z==63 -> 2).
__device__ inline int mask_cnt(int zi, int yi, int xi, int t) {
  int dz = (t >> 2) & 1, dy = (t >> 1) & 1, dx = t & 1;
  int rz = (zi == 31) ? (dz ? 2 : 1) : 0;
  int ry = (yi == 31) ? (dy ? 2 : 1) : 0;
  int rx = (xi == 31) ? (dx ? 2 : 1) : 0;
  return rz * 9 + ry * 3 + rx;
}

// Pack a KxNc f32 weight matrix into per-lane WMMA B-fragment layout:
// dst[((nt*Kt+kt)*32+lane)*16 + e] = W[kt*32 + (e>>3)*16 + (lane>>4)*8 + (e&7)][nt*16 + (lane&15)]
__global__ void swin_pack_w(const float* __restrict__ src, _Float16* __restrict__ dst,
                            int Kt, int Nt, int Nc) {
  int tid = blockIdx.x * 256 + threadIdx.x;
  if (tid >= Nt * Kt * 512) return;
  int e = tid & 15;
  int lane = (tid >> 4) & 31;
  int fi = tid >> 9;
  int kt = fi % Kt, nt = fi / Kt;
  int k = kt * 32 + (e >> 3) * 16 + (lane >> 4) * 8 + (e & 7);
  int n = nt * 16 + (lane & 15);
  dst[tid] = (_Float16)src[k * Nc + n];
}

// LN1 fused with roll(-1,-1,-1) + window partition: one wave per output row.
__global__ void swin_ln1_window(const float* __restrict__ x, const float* __restrict__ g1,
                                const float* __restrict__ b1, _Float16* __restrict__ xw) {
  int wave = threadIdx.x >> 5;
  int lane = threadIdx.x & 31;
  long n = (long)blockIdx.x * 8 + wave;          // window-order row index
  int b  = (int)(n >> 18);
  int m  = (int)(n & (LTOK - 1));
  int zi = m >> 13, yi = (m >> 8) & 31, xi = (m >> 3) & 31;
  int dz = (m >> 2) & 1, dy = (m >> 1) & 1, dx = m & 1;
  int zs = (zi * 2 + dz + 1) & 63;               // roll -1: out[i] = in[i+1]
  int ys = (yi * 2 + dy + 1) & 63;
  int xs = (xi * 2 + dx + 1) & 63;
  long src = ((long)b * LTOK + ((zs * 64 + ys) * 64 + xs)) * 64;
  float2 v = *reinterpret_cast<const float2*>(x + src + 2 * lane);
  float s = v.x + v.y, ss = v.x * v.x + v.y * v.y;
  #pragma unroll
  for (int off = 16; off; off >>= 1) { s += __shfl_xor(s, off, 32); ss += __shfl_xor(ss, off, 32); }
  float mu  = s * (1.0f / 64.0f);
  float var = ss * (1.0f / 64.0f) - mu * mu;
  float inv = rsqrtf(var + 1e-5f);
  int c0 = 2 * lane;
  union { _Float16 h[2]; unsigned u; } p;
  p.h[0] = (_Float16)((v.x - mu) * inv * g1[c0] + b1[c0]);
  p.h[1] = (_Float16)((v.y - mu) * inv * g1[c0 + 1] + b1[c0 + 1]);
  *reinterpret_cast<unsigned*>(xw + n * 64 + c0) = p.u;
}

// Fused QKV GEMM + window attention + proj GEMM + (bug-faithful) window reverse
// + roll(+1) + residual scatter into d_out (= x1). One wave per 16-token tile.
__global__ void __launch_bounds__(128)
swin_attn_fused(const _Float16* __restrict__ xw, const _Float16* __restrict__ wq,
                const float* __restrict__ qkv_b, const float* __restrict__ rpb,
                const _Float16* __restrict__ wp, const float* __restrict__ proj_b,
                const float* __restrict__ x, float* __restrict__ out) {
  __shared__ float    qkv_s[4][16 * 192];
  __shared__ _Float16 ao_s[4][16 * 64];
  int wid  = threadIdx.x >> 5;
  int lane = threadIdx.x & 31;
  int g    = lane >> 4;
  int mrow = lane & 15;
  int tile = blockIdx.x * 4 + wid;
  int row0 = tile * 16;

  // ---- phase 1: QKV = xw_tile(16x64) @ Wqkv(64x192) + b ----
  const _Float16* arow = xw + (long)(row0 + mrow) * 64 + g * 8;
  v16h a0 = load_frag16(arow);        // K 0..31
  v16h a1 = load_frag16(arow + 32);   // K 32..63
  #pragma unroll
  for (int nt = 0; nt < 12; ++nt) {
    v8f acc = {};
    acc = wmma32(a0, *reinterpret_cast<const v16h*>(wq + ((nt * 2 + 0) * 32 + lane) * 16), acc);
    acc = wmma32(a1, *reinterpret_cast<const v16h*>(wq + ((nt * 2 + 1) * 32 + lane) * 16), acc);
    int col = nt * 16 + mrow;
    float bv = qkv_b[col];
    #pragma unroll
    for (int r = 0; r < 8; ++r)
      qkv_s[wid][(r + 8 * g) * 192 + col] = acc[r] + bv;
  }
  __syncthreads();

  // ---- phase 2: attention over 2 windows x 8 heads x 8 queries = 128 tasks ----
  int wing = tile * 2;
  #pragma unroll
  for (int i = 0; i < 4; ++i) {
    int idx = i * 32 + lane;
    int w = idx >> 6, h = (idx >> 3) & 7, qt = idx & 7;
    int wm = (wing + w) & 32767;
    int zi = (wm >> 10) & 31, yi = (wm >> 5) & 31, xi = wm & 31;
    const float* base = &qkv_s[wid][(w * 8) * 192 + h * 8];
    float q[8];
    #pragma unroll
    for (int d = 0; d < 8; ++d) q[d] = base[qt * 192 + d];
    int cq = mask_cnt(zi, yi, xi, qt);
    int bi = rel_idx(h, qt) * 8;   // bug-faithful: bias[h,q,k] = rpb[IDX[h][q]][k]
    float sc[8]; float mx = -1e30f;
    #pragma unroll
    for (int j = 0; j < 8; ++j) {
      float dt = 0.f;
      #pragma unroll
      for (int d = 0; d < 8; ++d) dt += q[d] * base[j * 192 + 64 + d];
      dt = dt * 0.35355339059327379f + rpb[bi + j];
      if (mask_cnt(zi, yi, xi, j) != cq) dt -= 100.f;
      sc[j] = dt; mx = fmaxf(mx, dt);
    }
    float se = 0.f;
    #pragma unroll
    for (int j = 0; j < 8; ++j) { sc[j] = __expf(sc[j] - mx); se += sc[j]; }
    float rs = 1.f / se;
    #pragma unroll
    for (int d = 0; d < 8; ++d) {
      float o = 0.f;
      #pragma unroll
      for (int j = 0; j < 8; ++j) o += sc[j] * base[j * 192 + 128 + d];
      ao_s[wid][(w * 8 + qt) * 64 + h * 8 + d] = (_Float16)(o * rs);
    }
  }
  __syncthreads();

  // ---- phase 3: proj(64x64) + reverse-permute + roll(+1) + residual ----
  const _Float16* aob = &ao_s[wid][mrow * 64 + g * 8];
  v16h pa0 = load_frag16(aob);
  v16h pa1 = load_frag16(aob + 32);
  #pragma unroll
  for (int nt = 0; nt < 4; ++nt) {
    v8f acc = {};
    acc = wmma32(pa0, *reinterpret_cast<const v16h*>(wp + ((nt * 2 + 0) * 32 + lane) * 16), acc);
    acc = wmma32(pa1, *reinterpret_cast<const v16h*>(wp + ((nt * 2 + 1) * 32 + lane) * 16), acc);
    int col = nt * 16 + mrow;
    float bv = proj_b[col];
    #pragma unroll
    for (int r = 0; r < 8; ++r) {
      int trow = row0 + r + 8 * g;
      int b  = trow >> 18;
      int m  = trow & (LTOK - 1);
      // bug-faithful reverse reshape dims (32,2,32,32,2,2)
      int A1 = m >> 13, A2 = (m >> 12) & 1, A3 = (m >> 7) & 31;
      int A4 = (m >> 2) & 31, A5 = (m >> 1) & 1, A6 = m & 1;
      int zp = A1 * 2 + (A3 >> 4);
      int yp = (A3 & 15) * 4 + A5 * 2 + A2;
      int xp = A4 * 2 + A6;
      int zf = (zp + 1) & 63, yf = (yp + 1) & 63, xf = (xp + 1) & 63;  // roll +1
      long loc = ((long)b * LTOK + ((zf * 64 + yf) * 64 + xf)) * 64 + col;
      out[loc] = x[loc] + acc[r] + bv;
    }
  }
}

// Fused LN2 + MLP (64->256 GELU -> 256->64) + residual, in-place on d_out (x1).
__global__ void __launch_bounds__(128)
swin_mlp_fused(float* __restrict__ out, const float* __restrict__ g2, const float* __restrict__ b2,
               const _Float16* __restrict__ w1, const float* __restrict__ bb1,
               const _Float16* __restrict__ w2, const float* __restrict__ bb2) {
  __shared__ _Float16 ln_s[4][16 * 64];
  __shared__ _Float16 h_s[4][16 * 256];
  int wid  = threadIdx.x >> 5;
  int lane = threadIdx.x & 31;
  int g    = lane >> 4;
  int mrow = lane & 15;
  long row0 = (long)(blockIdx.x * 4 + wid) * 16;

  for (int r = 0; r < 16; ++r) {
    float2 v = *reinterpret_cast<const float2*>(out + (row0 + r) * 64 + 2 * lane);
    float s = v.x + v.y, ss = v.x * v.x + v.y * v.y;
    #pragma unroll
    for (int off = 16; off; off >>= 1) { s += __shfl_xor(s, off, 32); ss += __shfl_xor(ss, off, 32); }
    float mu  = s * (1.f / 64.f);
    float var = ss * (1.f / 64.f) - mu * mu;
    float inv = rsqrtf(var + 1e-5f);
    int c0 = 2 * lane;
    union { _Float16 h[2]; unsigned u; } p;
    p.h[0] = (_Float16)((v.x - mu) * inv * g2[c0] + b2[c0]);
    p.h[1] = (_Float16)((v.y - mu) * inv * g2[c0 + 1] + b2[c0 + 1]);
    *reinterpret_cast<unsigned*>(&ln_s[wid][r * 64 + c0]) = p.u;
  }
  __syncthreads();

  const _Float16* ab = &ln_s[wid][mrow * 64 + g * 8];
  v16h a0 = load_frag16(ab);
  v16h a1 = load_frag16(ab + 32);
  #pragma unroll
  for (int nt = 0; nt < 16; ++nt) {
    v8f acc = {};
    acc = wmma32(a0, *reinterpret_cast<const v16h*>(w1 + ((nt * 2 + 0) * 32 + lane) * 16), acc);
    acc = wmma32(a1, *reinterpret_cast<const v16h*>(w1 + ((nt * 2 + 1) * 32 + lane) * 16), acc);
    int col = nt * 16 + mrow;
    float bv = bb1[col];
    #pragma unroll
    for (int r = 0; r < 8; ++r) {
      float xv = acc[r] + bv;
      float gl = 0.5f * xv * (1.0f + erff(xv * 0.70710678118654752f));  // exact GELU
      h_s[wid][(r + 8 * g) * 256 + col] = (_Float16)gl;
    }
  }
  __syncthreads();

  const _Float16* hb = &h_s[wid][mrow * 256 + g * 8];
  v16h af[8];
  #pragma unroll
  for (int kt = 0; kt < 8; ++kt) af[kt] = load_frag16(hb + kt * 32);
  #pragma unroll
  for (int nt = 0; nt < 4; ++nt) {
    v8f acc = {};
    #pragma unroll
    for (int kt = 0; kt < 8; ++kt)
      acc = wmma32(af[kt], *reinterpret_cast<const v16h*>(w2 + ((nt * 8 + kt) * 32 + lane) * 16), acc);
    int col = nt * 16 + mrow;
    float bv = bb2[col];
    #pragma unroll
    for (int r = 0; r < 8; ++r) {
      long o = (row0 + r + 8 * g) * 64 + col;
      out[o] = out[o] + acc[r] + bv;
    }
  }
}

extern "C" void kernel_launch(void* const* d_in, const int* in_sizes, int n_in,
                              void* d_out, int out_size, void* d_ws, size_t ws_size,
                              hipStream_t stream) {
  const float* x      = (const float*)d_in[0];
  const float* n1g    = (const float*)d_in[1];
  const float* n1b    = (const float*)d_in[2];
  const float* qkv_w  = (const float*)d_in[3];
  const float* qkv_b  = (const float*)d_in[4];
  const float* rpb    = (const float*)d_in[5];
  const float* proj_w = (const float*)d_in[6];
  const float* proj_b = (const float*)d_in[7];
  const float* n2g    = (const float*)d_in[8];
  const float* n2b    = (const float*)d_in[9];
  const float* mlp_w1 = (const float*)d_in[10];
  const float* mlp_b1 = (const float*)d_in[11];
  const float* mlp_w2 = (const float*)d_in[12];
  const float* mlp_b2 = (const float*)d_in[13];
  float* out = (float*)d_out;

  _Float16* qkvw_f = (_Float16*)d_ws;              // 12 nt x 2 kt x 512 halves
  _Float16* projw_f = qkvw_f + 12288;              //  4 x 2 x 512
  _Float16* mlp1_f  = projw_f + 4096;              // 16 x 2 x 512
  _Float16* mlp2_f  = mlp1_f + 16384;              //  4 x 8 x 512
  _Float16* xw      = mlp2_f + 16384;              // NTOK x 64 halves (~64 MB)

  swin_pack_w<<<(12288 + 255) / 256, 256, 0, stream>>>(qkv_w,  qkvw_f, 2, 12, 192);
  swin_pack_w<<<(4096  + 255) / 256, 256, 0, stream>>>(proj_w, projw_f, 2, 4, 64);
  swin_pack_w<<<(16384 + 255) / 256, 256, 0, stream>>>(mlp_w1, mlp1_f, 2, 16, 256);
  swin_pack_w<<<(16384 + 255) / 256, 256, 0, stream>>>(mlp_w2, mlp2_f, 8, 4, 64);

  swin_ln1_window<<<NTOK / 8, 256, 0, stream>>>(x, n1g, n1b, xw);
  swin_attn_fused<<<NTOK / 64, 128, 0, stream>>>(xw, qkvw_f, qkv_b, rpb, projw_f, proj_b, x, out);
  swin_mlp_fused<<<NTOK / 64, 128, 0, stream>>>(out, n2g, n2b, mlp1_f, mlp_b1, mlp2_f, mlp_b2);
}
